// HGCN_78529182040166
// MI455X (gfx1250) — matrix operand order repaired and checked by reading, not compiled, not run on hardware
//
#include <hip/hip_runtime.h>
#include <hip/hip_bf16.h>
#include <stdint.h>

#define DIMS 256
#define NNODES 20000
#define PDIM 32
#define MAXN 0.996f     /* (1 - 4e-3)/sqrt(c), c = 1 */
#define MINN 1e-15f

typedef __attribute__((ext_vector_type(16))) _Float16 v16h;
typedef __attribute__((ext_vector_type(8)))  float    v8f;

// ---------------- helpers ----------------

__device__ __forceinline__ float blkSum(float v, float* sh) {
  int t = threadIdx.x;
  sh[t] = v;
  __syncthreads();
  #pragma unroll
  for (int o = 128; o > 0; o >>= 1) {
    if (t < o) sh[t] += sh[t + o];
    __syncthreads();
  }
  float r = sh[0];
  __syncthreads();
  return r;
}

__device__ __forceinline__ float artanhf_c(float x) {
  x = fminf(fmaxf(x, -0.99999f), 0.99999f);   // clip to +-(1 - 1e-5)
  return 0.5f * logf((1.f + x) / (1.f - x));
}

// ---------------- prep kernels ----------------

__global__ void k_f32_to_f16(const float* __restrict__ src, _Float16* __restrict__ dst, int n) {
  int i = blockIdx.x * 256 + threadIdx.x;
  if (i < n) dst[i] = (_Float16)src[i];
}

// hb = proj(expmap0(b)); also store ||hb||^2
__global__ void k_bias(const float* __restrict__ b, float* __restrict__ hb,
                       float* __restrict__ hbn2) {
  __shared__ float sh[256];
  int t = threadIdx.x;
  float v = b[t];
  float n2 = blkSum(v * v, sh);
  float nn = fmaxf(sqrtf(n2), MINN);
  float th = tanhf(nn);
  float h  = th * v / nn;
  float hn = th;
  float s = (hn > MAXN) ? (MAXN / hn) : 1.f;
  h *= s; hn = fminf(hn, MAXN);
  hb[t] = h;
  if (t == 0) hbn2[0] = hn * hn;
}

// tbl[v] = hyp_linear(proj(expmap0(emb[v])), Ws, bias) for each vocab row v
__global__ void k_table(const float* __restrict__ emb, const float* __restrict__ Ws,
                        const float* __restrict__ hb, const float* __restrict__ hbn2,
                        float* __restrict__ tbl, float* __restrict__ tbln2) {
  __shared__ float sh[256];
  __shared__ float p[PDIM];
  int v = blockIdx.x, t = threadIdx.x;
  float e = (t < PDIM) ? emb[v * PDIM + t] : 0.f;
  float n2 = blkSum(e * e, sh);
  float nn = fmaxf(sqrtf(n2), MINN);
  float th = tanhf(nn);
  float pe = th * e / nn;
  float pn = th;
  float s = (pn > MAXN) ? MAXN / pn : 1.f;
  if (t < PDIM) p[t] = pe * s;
  pn = fminf(pn, MAXN);
  float xn = fmaxf(pn, MINN);
  __syncthreads();
  float mv = 0.f;
  #pragma unroll
  for (int k = 0; k < PDIM; ++k) mv += p[k] * Ws[t * PDIM + k];
  float mvn2 = blkSum(mv * mv, sh);
  float mvn = fmaxf(sqrtf(mvn2), MINN);
  float tv = tanhf(mvn / xn * artanhf_c(xn));
  float res = tv * mv / mvn;              // ||res|| == tv (scalar)
  float rn = tv;
  s = (rn > MAXN) ? MAXN / rn : 1.f;
  res *= s; rn = fminf(rn, MAXN);
  float y2 = hbn2[0];
  float hv = hb[t];
  float xy = blkSum(res * hv, sh);
  float x2 = rn * rn;
  float den = fmaxf(1.f + 2.f * xy + x2 * y2, MINN);
  float o = ((1.f + 2.f * xy + y2) * res + (1.f - x2) * hv) / den;
  float on2 = blkSum(o * o, sh);
  float on = fmaxf(sqrtf(on2), MINN);
  s = (on > MAXN) ? MAXN / on : 1.f;
  o *= s; on = fminf(on, MAXN);
  tbl[v * DIMS + t] = o;
  if (t == 0) tbln2[v] = on * on;
}

// x = proj(expmap0(features)) -> f16 copy + norm
__global__ void k_init(const float* __restrict__ feat, _Float16* __restrict__ xf16,
                       float* __restrict__ xn) {
  __shared__ float sh[256];
  int n = blockIdx.x, t = threadIdx.x;
  float f = feat[(size_t)n * DIMS + t];
  float n2 = blkSum(f * f, sh);
  float nn = fmaxf(sqrtf(n2), MINN);
  float th = tanhf(nn);
  float x = th * f / nn;
  float xo = th;
  float s = (xo > MAXN) ? MAXN / xo : 1.f;
  x *= s; xo = fminf(xo, MAXN);
  xf16[(size_t)n * DIMS + t] = (_Float16)x;
  if (t == 0) xn[n] = fmaxf(xo, MINN);
}

// ---------------- WMMA GEMM: OUT[20000,256] = A[20000,256](f16) @ W[256,256]^T ----------------
// One wave per 16x16 output tile, K loop in steps of 32 via v_wmma_f32_16x16x32_f16.
__global__ void __launch_bounds__(256) k_gemm(const _Float16* __restrict__ A,
                                              const _Float16* __restrict__ W,
                                              float* __restrict__ OUT) {
  int wave = threadIdx.x >> 5;
  int lane = threadIdx.x & 31;
  int tile = blockIdx.x * 8 + wave;          // 0 .. 19999
  int mt = tile >> 4;                        // row tile (0..1249), N exact multiple of 16
  int nt = tile & 15;                        // col tile (0..15)
  int r  = lane & 15;
  int hi = lane >> 4;
  const _Float16* arow = A + ((size_t)(mt * 16 + r)) * DIMS;   // A row for this lane
  const _Float16* brow = W + ((size_t)(nt * 16 + r)) * DIMS;   // W row = B column
  v8f acc = {};
  #pragma unroll 2
  for (int kk = 0; kk < DIMS; kk += 32) {
    union { uint4 u[2]; v16h h; } ua, ub;
    // A 16x32 f16 frag: lanes 0-15 K={kk..kk+7, kk+16..kk+23}, lanes 16-31 shifted by 8
    ua.u[0] = *(const uint4*)(arow + kk + hi * 8);
    ua.u[1] = *(const uint4*)(arow + kk + 16 + hi * 8);
    // B 32x16 f16 frag: lane = column, K same as A: {kk..kk+7, kk+16..kk+23}, lanes 16-31 shifted by 8
    ub.u[0] = *(const uint4*)(brow + kk + hi * 8);
    ub.u[1] = *(const uint4*)(brow + kk + 16 + hi * 8);
    acc = __builtin_amdgcn_wmma_f32_16x16x32_f16(false, ua.h, false, ub.h,
                                                 (short)0, acc, false, false);
  }
  // D layout: lane l, VGPR i -> row = i + 8*(l>>4), col = l&15
  float* orow = OUT + (size_t)(mt * 16 + 8 * hi) * DIMS + nt * 16 + r;
  #pragma unroll
  for (int i = 0; i < 8; ++i) orow[(size_t)i * DIMS] = acc[i];
}

// ---------------- fused HyperbolicConcat (per node) ----------------
// In: MX = x @ Wa^T, XN = ||x||. Out: new_in (f16) + ||new_in||.
__global__ void k_concat(const float* __restrict__ MX, const float* __restrict__ XN,
                         const float* __restrict__ hba, const float* __restrict__ hban2,
                         const float* __restrict__ tblb, const float* __restrict__ tbbn2,
                         const float* __restrict__ tblc, const float* __restrict__ tbcn2,
                         const int* __restrict__ pos, const int* __restrict__ posa,
                         _Float16* __restrict__ xf16, float* __restrict__ xnout) {
  __shared__ float sh[256];
  int n = blockIdx.x, t = threadIdx.x;
  float mx = MX[(size_t)n * DIMS + t];
  float xn = XN[n];
  float mxn2 = blkSum(mx * mx, sh);
  float mxn = fmaxf(sqrtf(mxn2), MINN);
  float tv = tanhf(mxn / xn * artanhf_c(xn));
  float res = tv * mx / mxn;
  float rn = tv;
  float s = (rn > MAXN) ? MAXN / rn : 1.f;
  res *= s; rn = fminf(rn, MAXN);
  // mobius_add(res, hb(ba)) then proj  -> hA = hyp_linear(x, Wa, ba)
  float hv = hba[t]; float y2 = hban2[0];
  float xy = blkSum(res * hv, sh);
  float x2 = rn * rn;
  float den = fmaxf(1.f + 2.f * xy + x2 * y2, MINN);
  float h = ((1.f + 2.f * xy + y2) * res + (1.f - x2) * hv) / den;
  float hn2 = blkSum(h * h, sh);
  float hn = fmaxf(sqrtf(hn2), MINN);
  s = (hn > MAXN) ? MAXN / hn : 1.f;
  h *= s; hn = fminf(hn, MAXN);
  // mobius_add(hA, table_b[pos])
  int pv = pos[n];
  float tb = tblb[pv * DIMS + t]; y2 = tbbn2[pv];
  xy = blkSum(h * tb, sh);
  x2 = hn * hn;
  den = fmaxf(1.f + 2.f * xy + x2 * y2, MINN);
  float o1 = ((1.f + 2.f * xy + y2) * h + (1.f - x2) * tb) / den;
  float o1n2 = blkSum(o1 * o1, sh);
  // mobius_add(o1, table_c[posa]) (no proj between the two adds)
  int pa = posa[n];
  float tc = tblc[pa * DIMS + t]; y2 = tbcn2[pa];
  xy = blkSum(o1 * tc, sh);
  den = fmaxf(1.f + 2.f * xy + o1n2 * y2, MINN);
  float o2 = ((1.f + 2.f * xy + y2) * o1 + (1.f - o1n2) * tc) / den;
  float o2n2 = blkSum(o2 * o2, sh);
  float o2n = fmaxf(sqrtf(o2n2), MINN);
  s = (o2n > MAXN) ? MAXN / o2n : 1.f;
  o2 *= s; o2n = fminf(o2n, MAXN);
  xf16[(size_t)n * DIMS + t] = (_Float16)o2;
  if (t == 0) xnout[n] = fmaxf(o2n, MINN);
}

// ---------------- HypLinear(W,b) epilogue + logmap0, zero accumulator ----------------
__global__ void k_postlin(const float* __restrict__ MX, const float* __restrict__ XN,
                          const float* __restrict__ hbw, const float* __restrict__ hbwn2,
                          float* __restrict__ XT, float* __restrict__ SUP) {
  __shared__ float sh[256];
  int n = blockIdx.x, t = threadIdx.x;
  size_t idx = (size_t)n * DIMS + t;
  float mx = MX[idx];
  float xn = XN[n];
  float mxn2 = blkSum(mx * mx, sh);
  float mxn = fmaxf(sqrtf(mxn2), MINN);
  float tv = tanhf(mxn / xn * artanhf_c(xn));
  float res = tv * mx / mxn;
  float rn = tv;
  float s = (rn > MAXN) ? MAXN / rn : 1.f;
  res *= s; rn = fminf(rn, MAXN);
  float hv = hbw[t]; float y2 = hbwn2[0];
  float xy = blkSum(res * hv, sh);
  float x2 = rn * rn;
  float den = fmaxf(1.f + 2.f * xy + x2 * y2, MINN);
  float h = ((1.f + 2.f * xy + y2) * res + (1.f - x2) * hv) / den;
  float hn2 = blkSum(h * h, sh);
  float hn = fmaxf(sqrtf(hn2), MINN);
  s = (hn > MAXN) ? MAXN / hn : 1.f;
  h *= s; hn = fminf(hn, MAXN);
  float ln = fmaxf(hn, MINN);
  XT[idx] = artanhf_c(ln) / ln * h;   // logmap0
  SUP[idx] = 0.f;                     // zero the aggregation buffer
}

// ---------------- edge aggregation: SUP[dst] += XT[src] (64 threads x float4 per edge) ----------------
__global__ void k_edges(const int* __restrict__ src, const int* __restrict__ dst,
                        const float* __restrict__ xt, float* __restrict__ sup, int E) {
  int tid = blockIdx.x * 256 + threadIdx.x;
  int e = tid >> 6;
  if (e >= E) return;
  int dg = (tid & 63) * 4;
  int sN = src[e], dN = dst[e];
  const float4 v = *(const float4*)(xt + (size_t)sN * DIMS + dg);
  float* o = sup + (size_t)dN * DIMS + dg;
  __hip_atomic_fetch_add(o + 0, v.x, __ATOMIC_RELAXED, __HIP_MEMORY_SCOPE_AGENT);
  __hip_atomic_fetch_add(o + 1, v.y, __ATOMIC_RELAXED, __HIP_MEMORY_SCOPE_AGENT);
  __hip_atomic_fetch_add(o + 2, v.z, __ATOMIC_RELAXED, __HIP_MEMORY_SCOPE_AGENT);
  __hip_atomic_fetch_add(o + 3, v.w, __ATOMIC_RELAXED, __HIP_MEMORY_SCOPE_AGENT);
}

// ---------------- HypAgg tail + HypAct; final iteration emits logmap0 to d_out ----------------
__global__ void k_act(const float* __restrict__ SUP, _Float16* __restrict__ xf16,
                      float* __restrict__ xnout, float* __restrict__ outf, int final_out) {
  __shared__ float sh[256];
  int n = blockIdx.x, t = threadIdx.x;
  float sv = SUP[(size_t)n * DIMS + t];
  float n2 = blkSum(sv * sv, sh);
  float nn = fmaxf(sqrtf(n2), MINN);
  float th = tanhf(nn);
  float h = th * sv / nn;                   // expmap0
  float hn = th;
  float s = (hn > MAXN) ? MAXN / hn : 1.f;  // proj
  h *= s; hn = fminf(hn, MAXN);
  float ln = fmaxf(hn, MINN);
  float xt = fmaxf(artanhf_c(ln) / ln * h, 0.f);  // relu(logmap0)
  float r2 = blkSum(xt * xt, sh);
  float rn = fmaxf(sqrtf(r2), MINN);
  float th2 = tanhf(rn);
  float x = th2 * xt / rn;                  // expmap0
  float xo = th2;
  s = (xo > MAXN) ? MAXN / xo : 1.f;        // proj
  x *= s; xo = fminf(xo, MAXN);
  if (final_out) {
    float l2 = fmaxf(xo, MINN);
    outf[(size_t)n * DIMS + t] = artanhf_c(l2) / l2 * x;   // logmap0 -> output
  } else {
    xf16[(size_t)n * DIMS + t] = (_Float16)x;
    if (t == 0) xnout[n] = fmaxf(xo, MINN);
  }
}

// ---------------- host ----------------

extern "C" void kernel_launch(void* const* d_in, const int* in_sizes, int n_in,
                              void* d_out, int out_size, void* d_ws, size_t ws_size,
                              hipStream_t stream) {
  (void)n_in; (void)out_size; (void)ws_size;
  const float* feat = (const float*)d_in[0];
  const int* esrc = (const int*)d_in[1];
  const int* edst = (const int*)d_in[2];
  const int* pos  = (const int*)d_in[3];
  const int* posa = (const int*)d_in[4];
  const float* rel  = (const float*)d_in[5];
  const float* abse = (const float*)d_in[6];
  const float* W[2]  = {(const float*)d_in[7],  (const float*)d_in[15]};
  const float* b[2]  = {(const float*)d_in[8],  (const float*)d_in[16]};
  const float* Wa[2] = {(const float*)d_in[9],  (const float*)d_in[17]};
  const float* ba[2] = {(const float*)d_in[10], (const float*)d_in[18]};
  const float* Wb[2] = {(const float*)d_in[11], (const float*)d_in[19]};
  const float* bb[2] = {(const float*)d_in[12], (const float*)d_in[20]};
  const float* Wc[2] = {(const float*)d_in[13], (const float*)d_in[21]};
  const float* bc[2] = {(const float*)d_in[14], (const float*)d_in[22]};
  const int E = in_sizes[1];

  char* ws = (char*)d_ws;
  size_t off = 0;
  auto alloc = [&](size_t bytes) -> char* {
    char* p = ws + off;
    off += (bytes + 255) & ~(size_t)255;
    return p;
  };
  _Float16* XF16 = (_Float16*)alloc((size_t)NNODES * DIMS * 2);
  float* XN  = (float*)alloc((size_t)NNODES * 4);
  float* MX  = (float*)alloc((size_t)NNODES * DIMS * 4);
  float* XT  = (float*)alloc((size_t)NNODES * DIMS * 4);
  float* SUP = (float*)alloc((size_t)NNODES * DIMS * 4);
  _Float16* Wf[2]; _Float16* Waf[2];
  float* HB[2][4]; float* HBN2[2][4];
  float* TBLB[2]; float* TBBN2[2]; float* TBLC[2]; float* TBCN2[2];
  for (int l = 0; l < 2; ++l) {
    Wf[l]  = (_Float16*)alloc((size_t)DIMS * DIMS * 2);
    Waf[l] = (_Float16*)alloc((size_t)DIMS * DIMS * 2);
    for (int j = 0; j < 4; ++j) {
      HB[l][j]   = (float*)alloc(DIMS * 4);
      HBN2[l][j] = (float*)alloc(4);
    }
    TBLB[l]  = (float*)alloc(6 * DIMS * 4);
    TBBN2[l] = (float*)alloc(6 * 4);
    TBLC[l]  = (float*)alloc(20 * DIMS * 4);
    TBCN2[l] = (float*)alloc(20 * 4);
  }

  // ---- prep: f16 weights, hyperbolic biases, vocab tables ----
  const int cvt_blocks = (DIMS * DIMS + 255) / 256;
  for (int l = 0; l < 2; ++l) {
    k_f32_to_f16<<<cvt_blocks, 256, 0, stream>>>(W[l],  Wf[l],  DIMS * DIMS);
    k_f32_to_f16<<<cvt_blocks, 256, 0, stream>>>(Wa[l], Waf[l], DIMS * DIMS);
    k_bias<<<1, 256, 0, stream>>>(b[l],  HB[l][0], HBN2[l][0]);
    k_bias<<<1, 256, 0, stream>>>(ba[l], HB[l][1], HBN2[l][1]);
    k_bias<<<1, 256, 0, stream>>>(bb[l], HB[l][2], HBN2[l][2]);
    k_bias<<<1, 256, 0, stream>>>(bc[l], HB[l][3], HBN2[l][3]);
    k_table<<<6, 256, 0, stream>>>(rel,  Wb[l], HB[l][2], HBN2[l][2], TBLB[l], TBBN2[l]);
    k_table<<<20, 256, 0, stream>>>(abse, Wc[l], HB[l][3], HBN2[l][3], TBLC[l], TBCN2[l]);
  }

  k_init<<<NNODES, 256, 0, stream>>>(feat, XF16, XN);

  const int gemm_blocks = NNODES / 16 * 16 / 8;   // 20000 tiles / 8 waves = 2500
  const int edge_blocks = (E * 64 + 255) / 256;
  for (int l = 0; l < 2; ++l) {
    k_gemm<<<gemm_blocks, 256, 0, stream>>>(XF16, Waf[l], MX);
    k_concat<<<NNODES, 256, 0, stream>>>(MX, XN, HB[l][1], HBN2[l][1],
                                         TBLB[l], TBBN2[l], TBLC[l], TBCN2[l],
                                         pos, posa, XF16, XN);
    k_gemm<<<gemm_blocks, 256, 0, stream>>>(XF16, Wf[l], MX);
    k_postlin<<<NNODES, 256, 0, stream>>>(MX, XN, HB[l][0], HBN2[l][0], XT, SUP);
    k_edges<<<edge_blocks, 256, 0, stream>>>(esrc, edst, XT, SUP, E);
    k_act<<<NNODES, 256, 0, stream>>>(SUP, XF16, XN, (float*)d_out, (l == 1) ? 1 : 0);
  }
}
